// WideAndDeepModel_91010357002413
// MI455X (gfx1250) — matrix operand order repaired and checked by reading, not compiled, not run on hardware
//
#include <hip/hip_runtime.h>

// ---------------------------------------------------------------------------
// Wide & Deep fused forward for MI455X (gfx1250, wave32, WMMA f16->f32).
//
// Pipeline per wave (16 batch rows):
//   h0[16x256] (f16, LDS)  = concat(user_emb, item_emb, genre, tag), K-pad 248->256
//   h1 = relu(h0 @ W0 + b0)   248(->256) x 256   WMMA 16x16x32 f16
//   h2 = relu(h1 @ W1 + b1)   256 x 128
//   h3 = relu(h2 @ W2 + b2)   128 x 64
//   wide[16x64] (f32 gather of wide_W rows) -> f16, packed next to h3
//   logits = [h3|wide] @ final_W + final_b   128 x 5 (N padded to 16), WMMA
//
// Weights are pre-swizzled (fp32 -> f16) into the exact B-fragment layout so
// each lane loads 32 contiguous bytes per fragment (L2-resident, ~212 KB).
// ---------------------------------------------------------------------------

typedef _Float16 v16h __attribute__((ext_vector_type(16)));
typedef _Float16 v8h  __attribute__((ext_vector_type(8)));
typedef float    v8f  __attribute__((ext_vector_type(8)));

#define NUM_USERS_C 1000000
#define BATCH_C     16384
#define HROW        272          // LDS row stride in halves (544 B, 16B-aligned, conflict-skewed)

union AFrag { v16h v; v8h h[2]; };

// ---------------------------------------------------------------------------
// Pre-swizzle: W[K x N] fp32 row-major -> f16 fragment layout.
// Storage index: (((kt*NT + nt)*32 + lane)*16 + e), with
//   lane = g*16 + l,  K = kt*32 + g*8 + (e&7) + (e>>3)*16,  N = nt*16 + l.   (OOB -> 0)
// ---------------------------------------------------------------------------
__global__ void wd_swizzle_w(const float* __restrict__ W, _Float16* __restrict__ out,
                             int K, int N, int KT, int NT) {
  int idx = blockIdx.x * blockDim.x + threadIdx.x;
  int total = KT * NT * 512;
  if (idx >= total) return;
  int e    = idx & 15;
  int lane = (idx >> 4) & 31;
  int tile = idx >> 9;
  int nt = tile % NT;
  int kt = tile / NT;
  int l = lane & 15;
  int g = lane >> 4;
  int k = kt * 32 + g * 8 + (e & 7) + (e >> 3) * 16;
  int n = nt * 16 + l;
  float v = (k < K && n < N) ? W[(size_t)k * N + n] : 0.0f;
  out[idx] = (_Float16)v;
}

// ---------------------------------------------------------------------------
// One MLP layer for this wave's 16-row tile: bufOut = relu(bufIn @ W + bias).
// A fragment per the ISA 16-bit A 16x32 layout: lane g*16+l holds row M=l,
// elements 0..7 = K in [kt*32+g*8, +7], elements 8..15 = K in [kt*32+g*8+16, +23].
// ---------------------------------------------------------------------------
template <int KT, int NT>
__device__ __forceinline__ void wd_gemm_relu(const _Float16* __restrict__ bufIn,
                                             const _Float16* __restrict__ wS,
                                             const float* __restrict__ bias,
                                             _Float16* __restrict__ bufOut,
                                             int lane, int l, int g) {
  for (int nt = 0; nt < NT; ++nt) {
    float bv = bias[nt * 16 + l];
    v8f c = {bv, bv, bv, bv, bv, bv, bv, bv};
#pragma unroll
    for (int kt = 0; kt < KT; ++kt) {
      AFrag a;
      const _Float16* arow = bufIn + l * HROW + kt * 32 + g * 8;
      a.h[0] = *(const v8h*)(arow);        // ds_load_b128
      a.h[1] = *(const v8h*)(arow + 16);   // ds_load_b128
      AFrag b;
      b.v = *(const v16h*)(wS + (((size_t)kt * NT + nt) * 32 + lane) * 16);
      c = __builtin_amdgcn_wmma_f32_16x16x32_f16(false, a.v, false, b.v,
                                                 (short)0, c, false, false);
    }
#pragma unroll
    for (int r = 0; r < 8; ++r) {
      float v = c[r] > 0.0f ? c[r] : 0.0f;                 // ReLU
      bufOut[(r + 8 * g) * HROW + nt * 16 + l] = (_Float16)v;
    }
  }
}

__global__ __launch_bounds__(64)
void wd_fused(const int* __restrict__ user, const int* __restrict__ item,
              const float* __restrict__ genre, const float* __restrict__ tag,
              const float* __restrict__ wideW, const float* __restrict__ wideB,
              const float* __restrict__ uemb, const float* __restrict__ iemb,
              const float* __restrict__ b0, const float* __restrict__ b1,
              const float* __restrict__ b2, const float* __restrict__ fb,
              const _Float16* __restrict__ w0s, const _Float16* __restrict__ w1s,
              const _Float16* __restrict__ w2s, const _Float16* __restrict__ fws,
              float* __restrict__ out) {
  __shared__ __align__(16) _Float16 lds[2][2][16 * HROW];  // [wave][pingpong]
  const int lane = threadIdx.x & 31;
  const int wave = threadIdx.x >> 5;
  const int l = lane & 15;
  const int g = lane >> 4;
  const int row0 = (blockIdx.x * 2 + wave) * 16;

  _Float16* bufA = lds[wave][0];
  _Float16* bufB = lds[wave][1];

  // ---- phase 0: build h0 (16 x 256 f16; cols 248..255 zero) in bufA --------
  for (int i = lane; i < 16 * 256; i += 32) {
    int r = i >> 8;
    int c = i & 255;
    int grow = row0 + r;
    float v;
    if (c < 64)       v = uemb[(size_t)user[grow] * 64 + c];
    else if (c < 128) v = iemb[(size_t)item[grow] * 64 + (c - 64)];
    else if (c < 148) v = genre[(size_t)grow * 20 + (c - 128)];
    else if (c < 248) v = tag[(size_t)grow * 100 + (c - 148)];
    else              v = 0.0f;
    bufA[r * HROW + c] = (_Float16)v;
  }
  // same-wave DS ops are in-order: producer/consumer needs no barrier.

  // ---- deep MLP -------------------------------------------------------------
  wd_gemm_relu<8, 16>(bufA, w0s, b0, bufB, lane, l, g);  // 248(->256) -> 256
  wd_gemm_relu<8,  8>(bufB, w1s, b1, bufA, lane, l, g);  // 256 -> 128
  wd_gemm_relu<4,  4>(bufA, w2s, b2, bufB, lane, l, g);  // 128 -> 64 (bufB cols 0..63)

  // ---- wide path: gather + add, f16 into bufB cols 64..127 ------------------
  for (int i = lane; i < 16 * 64; i += 32) {
    int r = i >> 6;
    int c = i & 63;
    int grow = row0 + r;
    size_t u  = (size_t)user[grow];
    size_t it = (size_t)item[grow];
    float v = wideW[u * 64 + c] + wideW[(NUM_USERS_C + it) * 64 + c] + wideB[c];
    bufB[r * HROW + 64 + c] = (_Float16)v;
  }

  // ---- final projection: [h3|wide](16x128) @ final_W(128x5, N-pad 16) -------
  {
    float bv = (l < 5) ? fb[l] : 0.0f;
    v8f c = {bv, bv, bv, bv, bv, bv, bv, bv};
#pragma unroll
    for (int kt = 0; kt < 4; ++kt) {
      AFrag a;
      const _Float16* arow = bufB + l * HROW + kt * 32 + g * 8;
      a.h[0] = *(const v8h*)(arow);
      a.h[1] = *(const v8h*)(arow + 16);
      AFrag b;
      b.v = *(const v16h*)(fws + (((size_t)kt) * 32 + lane) * 16);  // NT == 1
      c = __builtin_amdgcn_wmma_f32_16x16x32_f16(false, a.v, false, b.v,
                                                 (short)0, c, false, false);
    }
    if (l < 5) {
#pragma unroll
      for (int r = 0; r < 8; ++r)
        out[(size_t)(row0 + r + 8 * g) * 5 + l] = c[r];
    }
  }
}

// ---------------------------------------------------------------------------
// Host launch. Workspace layout (halves):
//   w0s @ 0       : 8*16*512 = 65536   (248->256 x 256)
//   w1s @ 65536   : 8* 8*512 = 32768   (256 x 128)
//   w2s @ 98304   : 4* 4*512 =  8192   (128 x 64)
//   fws @ 106496  : 4* 1*512 =  2048   (128 x 5 -> 16)
// Total 217 KB.
// ---------------------------------------------------------------------------
extern "C" void kernel_launch(void* const* d_in, const int* in_sizes, int n_in,
                              void* d_out, int out_size, void* d_ws, size_t ws_size,
                              hipStream_t stream) {
  const int*   user   = (const int*)d_in[0];
  const int*   item   = (const int*)d_in[1];
  const float* genre  = (const float*)d_in[2];
  const float* tag    = (const float*)d_in[3];
  const float* wideW  = (const float*)d_in[4];
  const float* wideB  = (const float*)d_in[5];
  const float* uemb   = (const float*)d_in[6];
  const float* iemb   = (const float*)d_in[7];
  const float* W0     = (const float*)d_in[8];
  const float* b0     = (const float*)d_in[9];
  const float* W1     = (const float*)d_in[10];
  const float* b1     = (const float*)d_in[11];
  const float* W2     = (const float*)d_in[12];
  const float* b2     = (const float*)d_in[13];
  const float* finalW = (const float*)d_in[14];
  const float* fb     = (const float*)d_in[15];

  _Float16* ws  = (_Float16*)d_ws;
  _Float16* w0s = ws;
  _Float16* w1s = ws + 65536;
  _Float16* w2s = ws + 98304;
  _Float16* fws = ws + 106496;

  wd_swizzle_w<<<(8 * 16 * 512) / 256, 256, 0, stream>>>(W0,     w0s, 248, 256, 8, 16);
  wd_swizzle_w<<<(8 *  8 * 512) / 256, 256, 0, stream>>>(W1,     w1s, 256, 128, 8,  8);
  wd_swizzle_w<<<(4 *  4 * 512) / 256, 256, 0, stream>>>(W2,     w2s, 128,  64, 4,  4);
  wd_swizzle_w<<<(4 *  1 * 512) / 256, 256, 0, stream>>>(finalW, fws, 128,   5, 4,  1);

  wd_fused<<<BATCH_C / 32, 64, 0, stream>>>(user, item, genre, tag, wideW, wideB,
                                            uemb, iemb, b0, b1, b2, fb,
                                            w0s, w1s, w2s, fws, (float*)d_out);
}
